// S7_25709674234236
// MI455X (gfx1250) — compile-verified, matrix operand order checked by reading, not run on hardware
//
#include <hip/hip_runtime.h>
#include <hip/hip_bf16.h>
#include <stddef.h>

// ---------------------------------------------------------------------------
// Problem constants (from reference)
// ---------------------------------------------------------------------------
#define LSEQ   8192
#define HDIM   1024
#define PDIM   512
#define CHUNK  128
#define NCHUNK (LSEQ / CHUNK)   // 64

#define MT1 4                   // M-tiles per wave, GEMM1
#define MT2 4                   // M-tiles per wave, GEMM2
#define NT2 2                   // N-tiles per wave, GEMM2

typedef __bf16 bf16;
typedef __attribute__((ext_vector_type(16))) __bf16 v16bf;
typedef __attribute__((ext_vector_type(8)))  __bf16 v8bf;
typedef __attribute__((ext_vector_type(4)))  __bf16 v4bf;
typedef __attribute__((ext_vector_type(8)))  float  v8f;
typedef __attribute__((ext_vector_type(4)))  float  v4f;

// ---------------------------------------------------------------------------
// WMMA bf16 fragment loader (wave32).
// Storage is row-major (rows, K). Per ISA 7.12.2 16-bit A layout:
//   lane&15 selects the row, lane>>4 selects the K-half,
//   VGPR0-3 hold K = khalf*8 .. khalf*8+7, VGPR4-7 hold K = 16+khalf*8 .. +7.
// B operand uses the identical pattern on (N,K)-major storage.
// ---------------------------------------------------------------------------
__device__ __forceinline__ v16bf load_frag(const bf16* __restrict__ base,
                                           int ld, int lane) {
  const int r  = lane & 15;
  const int kh = lane >> 4;
  const bf16* p = base + (size_t)r * ld + kh * 8;
  v8bf lo = *(const v8bf*)(p);        // K = kh*8 .. kh*8+7
  v8bf hi = *(const v8bf*)(p + 16);   // K = 16 + kh*8 .. +7
  return __builtin_shufflevector(lo, hi,
      0,1,2,3,4,5,6,7,8,9,10,11,12,13,14,15);
}

// ---------------------------------------------------------------------------
// K0a: fp32 -> bf16 convert, 4 elements/thread (b128 load, b64 store)
// ---------------------------------------------------------------------------
__global__ void k_cvt_bf16_v4(const float* __restrict__ in,
                              bf16* __restrict__ out, int n4) {
  int i = blockIdx.x * blockDim.x + threadIdx.x;
  if (i >= n4) return;
  v4f v = *(const v4f*)(in + (size_t)i * 4);
  v4bf o;
  o[0] = (bf16)v[0]; o[1] = (bf16)v[1]; o[2] = (bf16)v[2]; o[3] = (bf16)v[3];
  *(v4bf*)(out + (size_t)i * 4) = o;
}

// K0b: W_dt (H,P) -> Wt (P,H) bf16 via 32x32 LDS tile (coalesced both sides)
__global__ __launch_bounds__(256) void k_cvt_transpose(
    const float* __restrict__ in, bf16* __restrict__ out) {
  __shared__ float tile[32][33];
  const int tx = threadIdx.x & 31;
  const int ty = threadIdx.x >> 5;            // 0..7
  const int h0 = blockIdx.y * 32;
  const int p0 = blockIdx.x * 32;
#pragma unroll
  for (int k = 0; k < 4; ++k) {
    int h = h0 + ty + k * 8;
    tile[ty + k * 8][tx] = in[(size_t)h * PDIM + p0 + tx];
  }
  __syncthreads();
#pragma unroll
  for (int k = 0; k < 4; ++k) {
    int p = p0 + ty + k * 8;
    out[(size_t)p * HDIM + h0 + tx] = (bf16)tile[tx][ty + k * 8];
  }
}

// ---------------------------------------------------------------------------
// K1: fused GEMM (u@W_dt, u@B_re^T, u@B_im^T) + ZOH discretization.
// 8 waves/block; wave owns MT1=4 M-tiles x 1 N-tile: B fragments loaded once
// per k-step, amortized over 12 WMMAs (4x reuse, 12 independent XDL chains).
// ---------------------------------------------------------------------------
__global__ __launch_bounds__(256) void k_gemm1_disc(
    const bf16*  __restrict__ u16,   // (L,H) bf16
    const bf16*  __restrict__ Wt,    // (P,H) bf16  (W_dt transposed)
    const bf16*  __restrict__ Bre,   // (P,H) bf16
    const bf16*  __restrict__ Bim,   // (P,H) bf16
    const float* __restrict__ bdt,   // (P)
    const float* __restrict__ intg,  // (L)
    const float* __restrict__ lamr,  // (P)
    const float* __restrict__ lami,  // (P)
    float* __restrict__ Are, float* __restrict__ Aim,
    float* __restrict__ BuR, float* __restrict__ BuI)
{
  const int lane   = threadIdx.x & 31;
  const int wave   = threadIdx.x >> 5;
  const int n0     = blockIdx.x * 16;                       // channel tile
  const int m_base = (blockIdx.y * 8 + wave) * (16 * MT1);  // sequence rows

  v8f accS[MT1] = {}; v8f accR[MT1] = {}; v8f accI[MT1] = {};
  const bf16* aBase = u16 + (size_t)m_base * HDIM;
  const bf16* sBase = Wt  + (size_t)n0 * HDIM;
  const bf16* rBase = Bre + (size_t)n0 * HDIM;
  const bf16* iBase = Bim + (size_t)n0 * HDIM;

  for (int kt = 0; kt < HDIM; kt += 32) {
    if (kt + 128 < HDIM)
      __builtin_prefetch(aBase + (size_t)(lane & 15) * HDIM + kt + 128, 0, 3);
    v16bf bS = load_frag(sBase + kt, HDIM, lane);
    v16bf bR = load_frag(rBase + kt, HDIM, lane);
    v16bf bI = load_frag(iBase + kt, HDIM, lane);
#pragma unroll
    for (int mt = 0; mt < MT1; ++mt) {
      v16bf a = load_frag(aBase + (size_t)mt * 16 * HDIM + kt, HDIM, lane);
      accS[mt] = __builtin_amdgcn_wmma_f32_16x16x32_bf16(false, a, false, bS,
                                            (short)0, accS[mt], false, false);
      accR[mt] = __builtin_amdgcn_wmma_f32_16x16x32_bf16(false, a, false, bR,
                                            (short)0, accR[mt], false, false);
      accI[mt] = __builtin_amdgcn_wmma_f32_16x16x32_bf16(false, a, false, bI,
                                            (short)0, accI[mt], false, false);
    }
  }

  // C/D layout: VGPR i -> M = i + 8*(lane>=16), N = lane&15
  const int kh = lane >> 4;
  const int gp = n0 + (lane & 15);
  const float lr = lamr[gp], li = lami[gp], bb = bdt[gp];
  const float inv_den = 1.0f / (lr * lr + li * li);
#pragma unroll
  for (int mt = 0; mt < MT1; ++mt) {
#pragma unroll
    for (int i = 0; i < 8; ++i) {
      const int gl = m_base + mt * 16 + i + 8 * kh;
      float x = accS[mt][i] + bb;
      float s = (x > 20.0f) ? x : log1pf(__expf(x));     // softplus
      float Delta = s * intg[gl];
      float er = __expf(lr * Delta);
      float sn, cs;
      __sincosf(li * Delta, &sn, &cs);
      float ar = er * cs, ai = er * sn;                  // Lambda_bar
      float ar1 = ar - 1.0f;
      float gr = (ar1 * lr + ai * li) * inv_den;         // (Lb-1)/Lambda
      float gi = (ai * lr - ar1 * li) * inv_den;
      float br = accR[mt][i], bi = accI[mt][i];
      size_t o = (size_t)gl * PDIM + gp;
      Are[o] = ar; Aim[o] = ai;
      BuR[o] = gr * br - gi * bi;
      BuI[o] = gr * bi + gi * br;
    }
  }
}

// ---------------------------------------------------------------------------
// K2a: per-chunk local scan. Thread owns one (chunk, channel) pair.
// In place: Bu <- local scan x, A <- inclusive prefix product of A.
// ---------------------------------------------------------------------------
__global__ __launch_bounds__(256) void k_scan_local(
    float* __restrict__ Are, float* __restrict__ Aim,
    float* __restrict__ BuR, float* __restrict__ BuI,
    float* __restrict__ AaR, float* __restrict__ AaI,
    float* __restrict__ XaR, float* __restrict__ XaI)
{
  const int tid = blockIdx.x * blockDim.x + threadIdx.x;  // 0 .. NCHUNK*P-1
  const int p = tid & (PDIM - 1);
  const int c = tid >> 9;                                 // /PDIM
  size_t base = (size_t)c * CHUNK * PDIM + p;
  float xr = 0.f, xi = 0.f, pr = 1.f, pi = 0.f;
  for (int t = 0; t < CHUNK; ++t) {
    size_t idx = base + (size_t)t * PDIM;
    float ar = Are[idx], ai = Aim[idx];
    float br = BuR[idx], bi = BuI[idx];
    float nxr = ar * xr - ai * xi + br;
    float nxi = ar * xi + ai * xr + bi;
    xr = nxr; xi = nxi;
    float npr = ar * pr - ai * pi;
    float npi = ar * pi + ai * pr;
    pr = npr; pi = npi;
    BuR[idx] = xr; BuI[idx] = xi;     // local inclusive scan
    Are[idx] = pr; Aim[idx] = pi;     // prefix product
  }
  AaR[tid] = pr; AaI[tid] = pi;
  XaR[tid] = xr; XaI[tid] = xi;
}

// K2b: serial scan of chunk aggregates -> incoming state per chunk.
__global__ void k_scan_carry(
    const float* __restrict__ AaR, const float* __restrict__ AaI,
    const float* __restrict__ XaR, const float* __restrict__ XaI,
    float* __restrict__ XinR, float* __restrict__ XinI)
{
  const int p = threadIdx.x;            // 512 threads
  float cr = 0.f, ci = 0.f;
  for (int c = 0; c < NCHUNK; ++c) {
    int idx = c * PDIM + p;
    XinR[idx] = cr; XinI[idx] = ci;
    float ar = AaR[idx], ai = AaI[idx];
    float nr = ar * cr - ai * ci + XaR[idx];
    float ni = ar * ci + ai * cr + XaI[idx];
    cr = nr; ci = ni;
  }
}

// K2c: fixup x = x_local + prefix * x_in; emit bf16 (xs_re, -xs_im).
// 4 consecutive channels per thread: all loads b128, stores b64.
__global__ __launch_bounds__(256) void k_scan_fix(
    const float* __restrict__ Are, const float* __restrict__ Aim,
    const float* __restrict__ BuR, const float* __restrict__ BuI,
    const float* __restrict__ XinR, const float* __restrict__ XinI,
    bf16* __restrict__ xsr, bf16* __restrict__ xsin)
{
  const int t = blockIdx.x * blockDim.x + threadIdx.x;    // < L*P/4
  const size_t e0 = (size_t)t * 4;
  const int l  = (int)(e0 >> 9);
  const int p  = (int)(e0 & (PDIM - 1));
  const int ci = (l >> 7) * PDIM + p;
  v4f pr = *(const v4f*)(Are + e0);
  v4f pi = *(const v4f*)(Aim + e0);
  v4f br = *(const v4f*)(BuR + e0);
  v4f bi = *(const v4f*)(BuI + e0);
  v4f ir = *(const v4f*)(XinR + ci);
  v4f ii = *(const v4f*)(XinI + ci);
  v4f xr = br + pr * ir - pi * ii;
  v4f xi = bi + pr * ii + pi * ir;
  v4bf oR, oI;
  oR[0] = (bf16)xr[0]; oR[1] = (bf16)xr[1];
  oR[2] = (bf16)xr[2]; oR[3] = (bf16)xr[3];
  oI[0] = (bf16)(-xi[0]); oI[1] = (bf16)(-xi[1]);
  oI[2] = (bf16)(-xi[2]); oI[3] = (bf16)(-xi[3]);
  *(v4bf*)(xsr  + e0) = oR;
  *(v4bf*)(xsin + e0) = oI;                // pre-negated imag for GEMM2
}

// ---------------------------------------------------------------------------
// K3: y = 2*(xs_re@C_re^T + (-xs_im)@C_im^T) + u*D via WMMA.
// Wave owns MT2=4 M-tiles x NT2=2 N-tiles: both A and B fragments reused,
// 16 WMMAs per k-step over 8 independent accumulator chains.
// ---------------------------------------------------------------------------
__global__ __launch_bounds__(256) void k_gemm2_out(
    const bf16*  __restrict__ xsr,   // (L,P) bf16
    const bf16*  __restrict__ xsin,  // (L,P) bf16 (negated imag)
    const bf16*  __restrict__ Cre,   // (H,P) bf16
    const bf16*  __restrict__ Cim,   // (H,P) bf16
    const float* __restrict__ u,     // (L,H) fp32
    const float* __restrict__ Dsk,   // (H)
    float* __restrict__ y)           // (L,H)
{
  const int lane   = threadIdx.x & 31;
  const int wave   = threadIdx.x >> 5;
  const int n_base = blockIdx.x * (16 * NT2);               // H cols
  const int m_base = (blockIdx.y * 8 + wave) * (16 * MT2);  // L rows

  v8f acc[MT2][NT2] = {};
  const bf16* aR = xsr  + (size_t)m_base * PDIM;
  const bf16* aI = xsin + (size_t)m_base * PDIM;

  for (int kt = 0; kt < PDIM; kt += 32) {
    if (kt + 128 < PDIM)
      __builtin_prefetch(aR + (size_t)(lane & 15) * PDIM + kt + 128, 0, 3);
    v16bf bRe[NT2], bIm[NT2];
#pragma unroll
    for (int n = 0; n < NT2; ++n) {
      bRe[n] = load_frag(Cre + (size_t)(n_base + n * 16) * PDIM + kt, PDIM, lane);
      bIm[n] = load_frag(Cim + (size_t)(n_base + n * 16) * PDIM + kt, PDIM, lane);
    }
#pragma unroll
    for (int mt = 0; mt < MT2; ++mt) {
      v16bf a1 = load_frag(aR + (size_t)mt * 16 * PDIM + kt, PDIM, lane);
#pragma unroll
      for (int n = 0; n < NT2; ++n)
        acc[mt][n] = __builtin_amdgcn_wmma_f32_16x16x32_bf16(false, a1, false,
                                bRe[n], (short)0, acc[mt][n], false, false);
      v16bf a2 = load_frag(aI + (size_t)mt * 16 * PDIM + kt, PDIM, lane);
#pragma unroll
      for (int n = 0; n < NT2; ++n)
        acc[mt][n] = __builtin_amdgcn_wmma_f32_16x16x32_bf16(false, a2, false,
                                bIm[n], (short)0, acc[mt][n], false, false);
    }
  }

  const int kh = lane >> 4;
  const int nc = lane & 15;
#pragma unroll
  for (int n = 0; n < NT2; ++n) {
    const int gh = n_base + n * 16 + nc;
    const float d = Dsk[gh];
#pragma unroll
    for (int mt = 0; mt < MT2; ++mt) {
#pragma unroll
      for (int i = 0; i < 8; ++i) {
        const int gl = m_base + mt * 16 + i + 8 * kh;
        size_t o = (size_t)gl * HDIM + gh;
        y[o] = 2.0f * acc[mt][n][i] + u[o] * d;
      }
    }
  }
}

// ---------------------------------------------------------------------------
// Workspace layout (bytes), all 256B-aligned by construction
// ---------------------------------------------------------------------------
static constexpr size_t SZ_U16 = (size_t)LSEQ * HDIM * 2;    // u bf16
static constexpr size_t SZ_PH  = (size_t)PDIM * HDIM * 2;    // weight bf16
static constexpr size_t SZ_LP4 = (size_t)LSEQ * PDIM * 4;    // (L,P) fp32
static constexpr size_t SZ_AGG = (size_t)NCHUNK * PDIM * 4;  // aggregates
static constexpr size_t SZ_LP2 = (size_t)LSEQ * PDIM * 2;    // (L,P) bf16

static constexpr size_t OFF_U16  = 0;
static constexpr size_t OFF_WT   = OFF_U16  + SZ_U16;
static constexpr size_t OFF_BRE  = OFF_WT   + SZ_PH;
static constexpr size_t OFF_BIM  = OFF_BRE  + SZ_PH;
static constexpr size_t OFF_CRE  = OFF_BIM  + SZ_PH;
static constexpr size_t OFF_CIM  = OFF_CRE  + SZ_PH;
static constexpr size_t OFF_ARE  = OFF_CIM  + SZ_PH;
static constexpr size_t OFF_AIM  = OFF_ARE  + SZ_LP4;
static constexpr size_t OFF_BUR  = OFF_AIM  + SZ_LP4;
static constexpr size_t OFF_BUI  = OFF_BUR  + SZ_LP4;
static constexpr size_t OFF_AAR  = OFF_BUI  + SZ_LP4;
static constexpr size_t OFF_AAI  = OFF_AAR  + SZ_AGG;
static constexpr size_t OFF_XAR  = OFF_AAI  + SZ_AGG;
static constexpr size_t OFF_XAI  = OFF_XAR  + SZ_AGG;
static constexpr size_t OFF_XINR = OFF_XAI  + SZ_AGG;
static constexpr size_t OFF_XINI = OFF_XINR + SZ_AGG;
static constexpr size_t OFF_XSR  = OFF_XINI + SZ_AGG;
static constexpr size_t OFF_XSI  = OFF_XSR  + SZ_LP2;

extern "C" void kernel_launch(void* const* d_in, const int* in_sizes, int n_in,
                              void* d_out, int out_size, void* d_ws, size_t ws_size,
                              hipStream_t stream) {
  (void)in_sizes; (void)n_in; (void)out_size; (void)ws_size;

  const float* u    = (const float*)d_in[0];
  const float* intg = (const float*)d_in[1];
  const float* lamr = (const float*)d_in[2];
  const float* lami = (const float*)d_in[3];
  const float* B_re = (const float*)d_in[4];
  const float* B_im = (const float*)d_in[5];
  const float* C_re = (const float*)d_in[6];
  const float* C_im = (const float*)d_in[7];
  const float* W_dt = (const float*)d_in[8];
  const float* b_dt = (const float*)d_in[9];
  const float* Dsk  = (const float*)d_in[10];
  float* y = (float*)d_out;

  char* ws = (char*)d_ws;
  bf16*  u16  = (bf16*)(ws + OFF_U16);
  bf16*  Wt   = (bf16*)(ws + OFF_WT);
  bf16*  Bre  = (bf16*)(ws + OFF_BRE);
  bf16*  Bim  = (bf16*)(ws + OFF_BIM);
  bf16*  Cre  = (bf16*)(ws + OFF_CRE);
  bf16*  Cim  = (bf16*)(ws + OFF_CIM);
  float* Are  = (float*)(ws + OFF_ARE);
  float* Aim  = (float*)(ws + OFF_AIM);
  float* BuR  = (float*)(ws + OFF_BUR);
  float* BuI  = (float*)(ws + OFF_BUI);
  float* AaR  = (float*)(ws + OFF_AAR);
  float* AaI  = (float*)(ws + OFF_AAI);
  float* XaR  = (float*)(ws + OFF_XAR);
  float* XaI  = (float*)(ws + OFF_XAI);
  float* XinR = (float*)(ws + OFF_XINR);
  float* XinI = (float*)(ws + OFF_XINI);
  bf16*  xsr  = (bf16*)(ws + OFF_XSR);
  bf16*  xsin = (bf16*)(ws + OFF_XSI);

  // --- K0: precision/layout prep (vectorized) ---
  k_cvt_bf16_v4<<<(LSEQ * HDIM / 4) / 256, 256, 0, stream>>>(u, u16,
                                                             LSEQ * HDIM / 4);
  k_cvt_bf16_v4<<<(PDIM * HDIM / 4) / 256, 256, 0, stream>>>(B_re, Bre,
                                                             PDIM * HDIM / 4);
  k_cvt_bf16_v4<<<(PDIM * HDIM / 4) / 256, 256, 0, stream>>>(B_im, Bim,
                                                             PDIM * HDIM / 4);
  k_cvt_bf16_v4<<<(HDIM * PDIM / 4) / 256, 256, 0, stream>>>(C_re, Cre,
                                                             HDIM * PDIM / 4);
  k_cvt_bf16_v4<<<(HDIM * PDIM / 4) / 256, 256, 0, stream>>>(C_im, Cim,
                                                             HDIM * PDIM / 4);
  k_cvt_transpose<<<dim3(PDIM / 32, HDIM / 32), 256, 0, stream>>>(W_dt, Wt);

  // --- K1: fused triple GEMM + discretization ---
  k_gemm1_disc<<<dim3(PDIM / 16, LSEQ / (128 * MT1)), 256, 0, stream>>>(
      u16, Wt, Bre, Bim, b_dt, intg, lamr, lami, Are, Aim, BuR, BuI);

  // --- K2: chunked complex linear scan over L ---
  k_scan_local<<<(NCHUNK * PDIM) / 256, 256, 0, stream>>>(
      Are, Aim, BuR, BuI, AaR, AaI, XaR, XaI);
  k_scan_carry<<<1, PDIM, 0, stream>>>(AaR, AaI, XaR, XaI, XinR, XinI);
  k_scan_fix<<<(LSEQ * PDIM / 4) / 256, 256, 0, stream>>>(
      Are, Aim, BuR, BuI, XinR, XinI, xsr, xsin);

  // --- K3: output GEMM + skip connection ---
  k_gemm2_out<<<dim3(HDIM / (16 * NT2), LSEQ / (128 * MT2)), 256, 0, stream>>>(
      xsr, xsin, Cre, Cim, u, Dsk, y);
}
